// SolarSpringAttention_52183852646832
// MI455X (gfx1250) — matrix operand order repaired and checked by reading, not compiled, not run on hardware
//
#include <hip/hip_runtime.h>
#include <hip/hip_bf16.h>

#define LTOK 4096
#define DDIM 2048
#define NPOS 8

typedef __attribute__((ext_vector_type(16))) __bf16 v16bf;
typedef __attribute__((ext_vector_type(8)))  float  v8f;

__constant__ float c_posw[NPOS] = {0.95f, 0.9f, 0.85f, 0.5f, 0.4f, 0.2f, 0.15f, 0.05f};

// ---------------- workspace layout (bytes) ----------------
static constexpr size_t OFF_TMAX  = 0;                                     // int
static constexpr size_t OFF_STATS = 1024;                                  // 9 * LTOK floats
static constexpr size_t OFF_XB    = 262144;                                // LTOK*DDIM bf16
static constexpr size_t OFF_WVH   = OFF_XB  + (size_t)LTOK * DDIM * 2;     // DDIM*DDIM bf16
static constexpr size_t OFF_WOH   = OFF_WVH + (size_t)DDIM * DDIM * 2;     // DDIM*DDIM bf16
static constexpr size_t OFF_AB    = OFF_WOH + (size_t)DDIM * DDIM * 2;     // LTOK*LTOK bf16
static constexpr size_t OFF_VT    = OFF_AB  + (size_t)LTOK * LTOK * 2;     // DDIM*LTOK bf16 (V transposed)
static constexpr size_t OFF_AVB   = OFF_VT  + (size_t)LTOK * DDIM * 2;     // LTOK*DDIM bf16
// total ~97 MB

__device__ __forceinline__ unsigned short f2bf(float f) {
  unsigned int u = __float_as_uint(f);
  unsigned int r = (u + 0x7FFFu + ((u >> 16) & 1u)) >> 16;   // round-nearest-even
  return (unsigned short)r;
}
__device__ __forceinline__ float sigm(float x) { return 1.0f / (1.0f + __expf(-x)); }

// ---------------- async global -> LDS path (gfx1250), with safe fallback ----------------
#if defined(__HIP_DEVICE_COMPILE__) && __has_builtin(__builtin_amdgcn_global_load_async_to_lds_b128)
#define ASYNC_LDS 1
#else
#define ASYNC_LDS 0
#endif

typedef int v4i_ __attribute__((vector_size(16)));
typedef __attribute__((address_space(1))) v4i_ gv4i;   // prints as "int __vector(4) __device__"
typedef __attribute__((address_space(3))) v4i_ lv4i;   // LDS

__device__ __forceinline__ void copy16_to_lds(const unsigned short* g, unsigned short* l) {
#if ASYNC_LDS
  __builtin_amdgcn_global_load_async_to_lds_b128((gv4i*)g, (lv4i*)l, 0, 0);
#else
  *reinterpret_cast<uint4*>(l) = *reinterpret_cast<const uint4*>(g);
#endif
}
__device__ __forceinline__ void async_wait0() {
#if ASYNC_LDS
#if __has_builtin(__builtin_amdgcn_s_wait_asynccnt)
  __builtin_amdgcn_s_wait_asynccnt(0);
#else
  asm volatile("s_wait_asynccnt 0x0" ::: "memory");
#endif
#endif
}

// ---------------- token_pos.max() ----------------
__global__ void k_init_tmax(int* tmax) {
  if (blockIdx.x == 0 && threadIdx.x == 0) *tmax = -2147483647;
}
__global__ void k_tmax(const int* __restrict__ tp, int* tmax) {
  int i = blockIdx.x * blockDim.x + threadIdx.x;
  if (i < LTOK) atomicMax(tmax, tp[i]);
}

// ---------------- per-token stats + X -> bf16 ----------------
__global__ __launch_bounds__(256) void k_stats(
    const float* __restrict__ X, const float* __restrict__ sun,
    const float* __restrict__ kspr, const float* __restrict__ lam,
    const int* __restrict__ posi, const int* __restrict__ dep,
    const int* __restrict__ tp, const int* __restrict__ slot,
    const int* __restrict__ tmax, float* __restrict__ stats,
    unsigned short* __restrict__ Xb)
{
  __shared__ float red[256];
  int i = blockIdx.x;
  int t = threadIdx.x;
  float s = 0.f;
  for (int d = t; d < DDIM; d += 256) {
    float v = X[(size_t)i * DDIM + d];
    s += v * v;
    Xb[(size_t)i * DDIM + d] = f2bf(v);
  }
  red[t] = s; __syncthreads();
  for (int o = 128; o > 0; o >>= 1) { if (t < o) red[t] += red[t + o]; __syncthreads(); }
  if (t == 0) {
    int p = ((posi[i] % NPOS) + NPOS) % NPOS;
    float mass = sqrtf(red[0]) * c_posw[p] * (1.0f + sun[i]);
    float age  = (float)(*tmax - tp[i]);
    float conf = __expf(-fabsf(lam[p]) * age);
    stats[0 * LTOK + i] = mass;
    stats[1 * LTOK + i] = conf;
    stats[2 * LTOK + i] = 1.0f / fmaxf(conf, 0.01f);
    stats[3 * LTOK + i] = (conf < 0.15f) ? 1.0f : 0.0f;
    stats[4 * LTOK + i] = sigm(kspr[p]);
    stats[5 * LTOK + i] = (float)slot[i];
    stats[6 * LTOK + i] = (float)dep[i];
    stats[7 * LTOK + i] = (float)tp[i];
    stats[8 * LTOK + i] = (float)p;
  }
}

// ---------------- weight f32 -> bf16 ----------------
__global__ __launch_bounds__(256) void k_convw(
    const float* __restrict__ Wv, const float* __restrict__ Wo,
    unsigned short* __restrict__ WvH, unsigned short* __restrict__ WoH)
{
  size_t i = (size_t)blockIdx.x * blockDim.x + threadIdx.x;
  if (i < (size_t)DDIM * DDIM) { WvH[i] = f2bf(Wv[i]); WoH[i] = f2bf(Wo[i]); }
}

// ---------------- pairwise score matrix ----------------
__global__ __launch_bounds__(256) void k_scores(
    const float* __restrict__ stats, const float* __restrict__ Gmic,
    const float* __restrict__ Gmac, const float* __restrict__ Gneu,
    const float* __restrict__ omg, const float* __restrict__ rst,
    float* __restrict__ S)
{
  size_t idx = (size_t)blockIdx.x * 256 + threadIdx.x;
  int i = (int)(idx >> 12);       // / 4096
  int j = (int)(idx & 4095);
  const float* mass = stats;
  const float* conf = stats + 1 * LTOK;
  const float* comp = stats + 2 * LTOK;
  const float* coll = stats + 3 * LTOK;
  const float* spr  = stats + 4 * LTOK;
  const float* slo  = stats + 5 * LTOK;
  const float* dpf  = stats + 6 * LTOK;
  const float* tpf  = stats + 7 * LTOK;
  const float* pof  = stats + 8 * LTOK;

  float mi = mass[i], mj = mass[j];
  int pi = (int)pof[i], pj = (int)pof[j];
  float slot_d = fabsf(slo[i] - slo[j]);
  float dep_d  = fabsf(dpf[i] - dpf[j]);
  float tok_d  = fabsf(tpf[i] - tpf[j]);

  float Gs     = sigm(Gmic[pi * NPOS + pj]);
  float r_slot = fmaxf(slot_d, 1.0f);
  float Fmic   = Gs * mi * mj / (r_slot * r_slot);
  float r_orb  = dep_d + 1.0f;
  float Fmac   = sigm(Gmac[0]) * mi * mj / (r_orb * r_orb);
  float Fspr   = spr[i] * tok_d;
  float Fns    = fabsf(Gneu[0]) * comp[i] * comp[j] * mi * mj
                 / fmaxf(r_slot * r_slot, 0.1f) * (coll[i] * coll[j]);
  float r_c    = fmaxf(slot_d, 0.1f);
  float om     = omg[0];
  float Forb   = mi * mj / r_c - mi * (om * om) * r_c;
  float rs     = fabsf(rst[0]);
  float Flag   = fmaxf(1.0f - fabsf(tok_d - rs) / (rs + 1.0f), 0.0f);

  float sc = (Fmic + Fmac + Fspr + Fns + Forb + Flag) * (conf[i] * conf[j]);
  if (i == j) sc = 0.0f;
  S[idx] = sc;
}

// ---------------- row softmax: A (f32) + A (bf16) ----------------
__global__ __launch_bounds__(256) void k_softmax(
    const float* __restrict__ S, const float* __restrict__ temp,
    float* __restrict__ A, unsigned short* __restrict__ Ab)
{
  __shared__ float red[256];
  int row = blockIdx.x;
  int t = threadIdx.x;
  const float* s = S + (size_t)row * LTOK;
  float* a = A + (size_t)row * LTOK;
  unsigned short* ab = Ab + (size_t)row * LTOK;
  float invT = 1.0f / fmaxf(fabsf(temp[0]), 0.1f);

  float m = -3.4e38f;
  for (int j = t; j < LTOK; j += 256) m = fmaxf(m, s[j] * invT);
  red[t] = m; __syncthreads();
  for (int o = 128; o > 0; o >>= 1) { if (t < o) red[t] = fmaxf(red[t], red[t + o]); __syncthreads(); }
  m = red[0]; __syncthreads();

  float sum = 0.f;
  for (int j = t; j < LTOK; j += 256) { float e = __expf(s[j] * invT - m); a[j] = e; sum += e; }
  red[t] = sum; __syncthreads();
  for (int o = 128; o > 0; o >>= 1) { if (t < o) red[t] += red[t + o]; __syncthreads(); }
  float inv = 1.0f / red[0];

  for (int j = t; j < LTOK; j += 256) { float v = a[j] * inv; a[j] = v; ab[j] = f2bf(v); }
}

// ---------------- bf16 WMMA GEMM with async-LDS double buffering ----------------
// C(MxN) = A(MxK) * W(NxK)^T (+bias). Block tile 64x128 (8 waves, each 32x32 =
// 2x2 v_wmma_f32_16x16x32_bf16). K-step 32. A tile 64x32 (4KB) and B tile
// 128x32 (8KB) staged into LDS with GLOBAL_LOAD_ASYNC_TO_LDS_B128 (ASYNCcnt),
// double buffered: the async engine fills buffer k+1 while WMMAs consume
// buffer k via ds_load_b128.
//
// Fragment layouts per ISA 7.12.2 (lane: half=lane>>4, sub=lane&15):
//  A-frag: elems 0..7  <- A[m, k+8*half .. +7]; elems 8..15 <- A[m, k+16+8*half .. +7]
//  B-frag (B=W^T):      elems 0..7  <- W[n, k+8*half .. +7]; elems 8..15 <- W[n, k+16+8*half .. +7]
//  D: VGPR r -> (M = r + 8*half, N = sub)
struct bf16x16 { uint4 lo, hi; };

__global__ __launch_bounds__(256) void k_gemm_bf16(
    const unsigned short* __restrict__ Am,
    const unsigned short* __restrict__ Wm,
    const float* __restrict__ bias,
    float* __restrict__ outf,
    unsigned short* __restrict__ outb,
    int M, int N, int K, int transOutB)
{
  __shared__ __align__(16) unsigned short lA[2][64][32];
  __shared__ __align__(16) unsigned short lB[2][128][32];

  int t    = threadIdx.x;
  int lane = t & 31;
  int wave = t >> 5;
  int half = lane >> 4;
  int sub  = lane & 15;
  int wm   = wave >> 2;      // 0..1
  int wn   = wave & 3;       // 0..3
  int m0b  = blockIdx.y * 64;
  int n0b  = blockIdx.x * 128;

  auto issue = [&](int buf, int k) {
    // A tile: 64 rows x 32 cols = 256 16-byte chunks, one per thread
    int ar = t >> 2, ac = (t & 3) * 8;
    copy16_to_lds(Am + (size_t)(m0b + ar) * K + k + ac, &lA[buf][ar][ac]);
    // B tile: 128 rows x 32 cols = 512 chunks, two per thread
#pragma unroll
    for (int i = 0; i < 2; ++i) {
      int idx = t + i * 256;
      int br = idx >> 2, bc = (idx & 3) * 8;
      copy16_to_lds(Wm + (size_t)(n0b + br) * K + k + bc, &lB[buf][br][bc]);
    }
  };

  v8f zero = {0.f, 0.f, 0.f, 0.f, 0.f, 0.f, 0.f, 0.f};
  v8f c[2][2];
  c[0][0] = zero; c[0][1] = zero; c[1][0] = zero; c[1][1] = zero;

  const int KT = K >> 5;
  issue(0, 0);
  async_wait0();
  __syncthreads();

  for (int kt = 0; kt < KT; ++kt) {
    int cur = kt & 1;
    if (kt + 1 < KT) issue(cur ^ 1, (kt + 1) << 5);   // overlap with compute below

    const unsigned short* pa0 = &lA[cur][wm * 32 + sub][0];
    const unsigned short* pa1 = &lA[cur][wm * 32 + 16 + sub][0];
    const unsigned short* pb0 = &lB[cur][wn * 32 + sub][0];
    const unsigned short* pb1 = &lB[cur][wn * 32 + 16 + sub][0];

    bf16x16 ta0, ta1, tb0, tb1;
    ta0.lo = *reinterpret_cast<const uint4*>(pa0 + 8 * half);
    ta0.hi = *reinterpret_cast<const uint4*>(pa0 + 16 + 8 * half);
    ta1.lo = *reinterpret_cast<const uint4*>(pa1 + 8 * half);
    ta1.hi = *reinterpret_cast<const uint4*>(pa1 + 16 + 8 * half);
    tb0.lo = *reinterpret_cast<const uint4*>(pb0 + 8 * half);
    tb0.hi = *reinterpret_cast<const uint4*>(pb0 + 16 + 8 * half);
    tb1.lo = *reinterpret_cast<const uint4*>(pb1 + 8 * half);
    tb1.hi = *reinterpret_cast<const uint4*>(pb1 + 16 + 8 * half);
    v16bf a0 = __builtin_bit_cast(v16bf, ta0);
    v16bf a1 = __builtin_bit_cast(v16bf, ta1);
    v16bf b0 = __builtin_bit_cast(v16bf, tb0);
    v16bf b1 = __builtin_bit_cast(v16bf, tb1);

    c[0][0] = __builtin_amdgcn_wmma_f32_16x16x32_bf16(false, a0, false, b0, (short)0, c[0][0], false, false);
    c[0][1] = __builtin_amdgcn_wmma_f32_16x16x32_bf16(false, a0, false, b1, (short)0, c[0][1], false, false);
    c[1][0] = __builtin_amdgcn_wmma_f32_16x16x32_bf16(false, a1, false, b0, (short)0, c[1][0], false, false);
    c[1][1] = __builtin_amdgcn_wmma_f32_16x16x32_bf16(false, a1, false, b1, (short)0, c[1][1], false, false);

    if (kt + 1 < KT) {
      async_wait0();      // wave's async fills of buffer cur^1 complete
      __syncthreads();    // whole block's fills visible; done reading cur
    }
  }

  int m0 = m0b + wm * 32;
  int n0 = n0b + wn * 32;
  float bv0 = bias ? bias[n0 + sub]      : 0.0f;
  float bv1 = bias ? bias[n0 + 16 + sub] : 0.0f;

#pragma unroll
  for (int ta = 0; ta < 2; ta++) {
#pragma unroll
    for (int tb = 0; tb < 2; tb++) {
      float bv = tb ? bv1 : bv0;
      int ncol = n0 + tb * 16 + sub;
#pragma unroll
      for (int r = 0; r < 8; r++) {
        int mr = m0 + ta * 16 + r + 8 * half;
        float v = c[ta][tb][r] + bv;
        if (outf) outf[(size_t)mr * N + ncol] = v;
        if (outb) {
          if (transOutB) outb[(size_t)ncol * M + mr] = f2bf(v);
          else           outb[(size_t)mr * N + ncol] = f2bf(v);
        }
      }
    }
  }
}

extern "C" void kernel_launch(void* const* d_in, const int* in_sizes, int n_in,
                              void* d_out, int out_size, void* d_ws, size_t ws_size,
                              hipStream_t stream) {
  (void)in_sizes; (void)n_in; (void)out_size; (void)ws_size;

  const float* X    = (const float*)d_in[0];
  const float* sun  = (const float*)d_in[1];
  const float* Gmic = (const float*)d_in[2];
  const float* Gmac = (const float*)d_in[3];
  const float* kspr = (const float*)d_in[4];
  const float* temp = (const float*)d_in[5];
  const float* lam  = (const float*)d_in[6];
  const float* Gneu = (const float*)d_in[7];
  const float* omg  = (const float*)d_in[8];
  const float* rst  = (const float*)d_in[9];
  const float* Wv   = (const float*)d_in[10];
  const float* bWv  = (const float*)d_in[11];
  const float* Wo   = (const float*)d_in[12];
  const float* bWo  = (const float*)d_in[13];
  const int* posi   = (const int*)d_in[14];
  const int* dep    = (const int*)d_in[15];
  const int* tpos   = (const int*)d_in[16];
  const int* slot   = (const int*)d_in[17];

  float* out  = (float*)d_out;                     // (L, D)
  float* Aout = out + (size_t)LTOK * DDIM;         // (L, L)
  float* Sout = Aout + (size_t)LTOK * LTOK;        // (L, L)

  char* ws = (char*)d_ws;
  int* tmax           = (int*)(ws + OFF_TMAX);
  float* stats        = (float*)(ws + OFF_STATS);
  unsigned short* Xb  = (unsigned short*)(ws + OFF_XB);
  unsigned short* WvH = (unsigned short*)(ws + OFF_WVH);
  unsigned short* WoH = (unsigned short*)(ws + OFF_WOH);
  unsigned short* Ab  = (unsigned short*)(ws + OFF_AB);
  unsigned short* Vt  = (unsigned short*)(ws + OFF_VT);
  unsigned short* AVb = (unsigned short*)(ws + OFF_AVB);

  k_init_tmax<<<1, 1, 0, stream>>>(tmax);
  k_tmax<<<LTOK / 256, 256, 0, stream>>>(tpos, tmax);
  k_stats<<<LTOK, 256, 0, stream>>>(X, sun, kspr, lam, posi, dep, tpos, slot, tmax, stats, Xb);
  k_convw<<<(DDIM * DDIM) / 256, 256, 0, stream>>>(Wv, Wo, WvH, WoH);
  k_scores<<<(LTOK * (size_t)LTOK) / 256, 256, 0, stream>>>(stats, Gmic, Gmac, Gneu, omg, rst, Sout);
  k_softmax<<<LTOK, 256, 0, stream>>>(Sout, temp, Aout, Ab);

  dim3 g(DDIM / 128, LTOK / 64);
  // V^T = (X * Wv^T + bWv)^T, stored bf16 transposed so GEMM2's B operand is N x K row-major
  k_gemm_bf16<<<g, 256, 0, stream>>>(Xb, WvH, bWv, nullptr, Vt, LTOK, DDIM, DDIM, 1);
  // AV = A * V  (B = (V^T)^T), bf16 out row-major
  k_gemm_bf16<<<g, 256, 0, stream>>>(Ab, Vt, nullptr, nullptr, AVb, LTOK, DDIM, LTOK, 0);
  // out = AV * Wo^T + bWo, f32 out
  k_gemm_bf16<<<g, 256, 0, stream>>>(AVb, WoH, bWo, out, nullptr, LTOK, DDIM, DDIM, 0);
}
